// MoEMambaUpdater_33621003993992
// MI455X (gfx1250) — matrix operand order repaired and checked by reading, not compiled, not run on hardware
//
#include <hip/hip_runtime.h>
#include <cstdint>
#include <cstddef>

// ---- problem constants (from reference) ----
static constexpr int BB      = 8;
static constexpr int TT      = 2048;
static constexpr int MTOK    = BB * TT;      // 16384 rows
static constexpr int D_IN    = 1088;
static constexpr int DMODEL  = 256;
static constexpr int D_OUT   = 64;
static constexpr int LAYERS  = 2;
static constexpr int EXPERTS = 4;
static constexpr int D_INNER = 512;
static constexpr int D_STATE = 16;
static constexpr int D_CONV  = 4;
static constexpr int DT_RANK = 16;
static constexpr int D_FFN   = 1024;

typedef __attribute__((ext_vector_type(16))) __bf16 bf16x16;
typedef __attribute__((ext_vector_type(8)))  float  f32x8;

union FragU { bf16x16 v; uint4 q[2]; };

#define LDSK 48   // padded LDS row stride in halfs (32 data + 16 pad)

// ---------------------------------------------------------------------
// staging helpers: global f32 tile -> LDS bf16 tile (32 K-columns)
// ---------------------------------------------------------------------
__device__ __forceinline__ void stage_tile_fast(const float* __restrict__ G, int ld,
                                                int base, int k0, __bf16* lds,
                                                int tid, int iters /*4 for 128 rows, 2 for 64*/)
{
    #pragma unroll
    for (int i = 0; i < iters; ++i) {
        int idx = i * 256 + tid;
        int r   = idx >> 3;          // row in tile
        int c4  = (idx & 7) << 2;    // K column group of 4
        float4 f = *(const float4*)&G[(size_t)(base + r) * ld + k0 + c4];
        union { __bf16 h[4]; uint2 u; } pk;
        pk.h[0] = (__bf16)f.x; pk.h[1] = (__bf16)f.y;
        pk.h[2] = (__bf16)f.z; pk.h[3] = (__bf16)f.w;
        *(uint2*)&lds[r * LDSK + c4] = pk.u;
    }
}

__device__ __forceinline__ void stage_tile_slow(const float* __restrict__ G, int ld,
                                                int base, int lim, int k0, int Kd,
                                                __bf16* lds, int tid, int iters /*16 or 8*/)
{
    for (int i = 0; i < iters; ++i) {
        int idx = i * 256 + tid;
        int r = idx >> 5, c = idx & 31;
        int g = base + r, gk = k0 + c;
        float v = 0.0f;
        if (g < lim && gk < Kd) v = G[(size_t)g * ld + gk];
        lds[r * LDSK + c] = (__bf16)v;
    }
}

// =====================================================================
// Tiled WMMA GEMM:  C[M,N] (+)= scale[row] * (A[M,K] @ W[N,K]^T + bias[N])
// f32 operands in HBM, truncated to bf16 while staging into LDS (b128
// loads + cvt_pk + ds_store_b64 on the uniform fast path), double-
// buffered LDS, v_wmma_f32_16x16x32_bf16 with f32 accumulation.
// Block tile 128x64, 8 waves (wave32), 32x32 per wave (2x2 WMMA subtiles).
// =====================================================================
__launch_bounds__(256)
__global__ void gemm_bf16_wmma(const float* __restrict__ A, int lda,
                               const float* __restrict__ W, int ldw,
                               const float* __restrict__ bias,
                               const float* __restrict__ scale, int sstride,
                               float* __restrict__ C, int ldc,
                               int M, int N, int Kd, int accum)
{
    __shared__ __bf16 lA[2][128 * LDSK];
    __shared__ __bf16 lB[2][64  * LDSK];

    const int tid  = threadIdx.x;
    const int lane = tid & 31;
    const int wv   = tid >> 5;
    const int wr   = (wv & 3) * 32;   // wave row offset in tile
    const int wc   = (wv >> 2) * 32;  // wave col offset in tile
    const int rowBase = blockIdx.y * 128;
    const int colBase = blockIdx.x * 64;

    const bool fullRows = (rowBase + 128 <= M);   // block-uniform (SGPR) conditions
    const bool fullCols = (colBase + 64  <= N);
    const int  nK = (Kd + 31) >> 5;

    // stage chunk 0 into buffer 0
    {
        const bool fk = (32 <= Kd);
        if (fullRows && fk) stage_tile_fast(A, lda, rowBase, 0, lA[0], tid, 4);
        else                stage_tile_slow(A, lda, rowBase, M, 0, Kd, lA[0], tid, 16);
        if (fullCols && fk) stage_tile_fast(W, ldw, colBase, 0, lB[0], tid, 2);
        else                stage_tile_slow(W, ldw, colBase, N, 0, Kd, lB[0], tid, 8);
    }
    __syncthreads();

    f32x8 acc[2][2] = {};

    for (int kc = 0; kc < nK; ++kc) {
        const int cur = kc & 1;
        const int nxt = cur ^ 1;

        // prefetch chunk kc+2 (lowers to global_prefetch_b8)
        if (kc + 2 < nK) {
            int r = tid >> 1, half = tid & 1;
            int gr = rowBase + r;
            if (gr < M) __builtin_prefetch(&A[(size_t)gr * lda + (kc + 2) * 32 + half * 16], 0, 1);
        }
        // stage chunk kc+1 into the other buffer (overlaps with WMMAs below)
        if (kc + 1 < nK) {
            const int  k0n = (kc + 1) * 32;
            const bool fk  = (k0n + 32 <= Kd);
            if (fullRows && fk) stage_tile_fast(A, lda, rowBase, k0n, lA[nxt], tid, 4);
            else                stage_tile_slow(A, lda, rowBase, M, k0n, Kd, lA[nxt], tid, 16);
            if (fullCols && fk) stage_tile_fast(W, ldw, colBase, k0n, lB[nxt], tid, 2);
            else                stage_tile_slow(W, ldw, colBase, N, k0n, Kd, lB[nxt], tid, 8);
        }

        // A-fragment layout (16-bit A 16x32): lanes 0-15 row M=lane,
        //   VGPR0-3 = K 0..7, VGPR4-7 = K 16..23; lanes 16-31 same rows, K +8.
        // B-fragment layout (32x16, K-major per lane): lane n%16 = column,
        //   halfs 0..7 = K 0..7, halfs 8..15 = K 16..23; lanes 16-31 K +8.
        {
            const __bf16* bufA = lA[cur];
            const __bf16* bufB = lB[cur];
            const int rA = lane & 15;
            const int kA = (lane >> 4) * 8;
            const int kB = (lane >> 4) * 8;
            FragU a0, a1, b0, b1;
            a0.q[0] = *(const uint4*)&bufA[(wr + rA)      * LDSK + kA];
            a0.q[1] = *(const uint4*)&bufA[(wr + rA)      * LDSK + kA + 16];
            a1.q[0] = *(const uint4*)&bufA[(wr + 16 + rA) * LDSK + kA];
            a1.q[1] = *(const uint4*)&bufA[(wr + 16 + rA) * LDSK + kA + 16];
            b0.q[0] = *(const uint4*)&bufB[(wc + rA)      * LDSK + kB];
            b0.q[1] = *(const uint4*)&bufB[(wc + rA)      * LDSK + kB + 16];
            b1.q[0] = *(const uint4*)&bufB[(wc + 16 + rA) * LDSK + kB];
            b1.q[1] = *(const uint4*)&bufB[(wc + 16 + rA) * LDSK + kB + 16];

            acc[0][0] = __builtin_amdgcn_wmma_f32_16x16x32_bf16(false, a0.v, false, b0.v, (short)0, acc[0][0], false, false);
            acc[0][1] = __builtin_amdgcn_wmma_f32_16x16x32_bf16(false, a0.v, false, b1.v, (short)0, acc[0][1], false, false);
            acc[1][0] = __builtin_amdgcn_wmma_f32_16x16x32_bf16(false, a1.v, false, b0.v, (short)0, acc[1][0], false, false);
            acc[1][1] = __builtin_amdgcn_wmma_f32_16x16x32_bf16(false, a1.v, false, b1.v, (short)0, acc[1][1], false, false);
        }
        __syncthreads();
    }

    // Epilogue. D layout: lane l, VGPR v -> row = (l/16)*8 + v, col = l%16.
    if (fullRows && fullCols) {
        #pragma unroll
        for (int i = 0; i < 2; ++i) {
            #pragma unroll
            for (int j = 0; j < 2; ++j) {
                int col   = colBase + wc + j * 16 + (lane & 15);
                int rbase = rowBase + wr + i * 16 + (lane >> 4) * 8;
                float bv = bias ? bias[col] : 0.0f;
                #pragma unroll
                for (int v = 0; v < 8; ++v) {
                    int rr = rbase + v;
                    float val = acc[i][j][v] + bv;
                    if (scale) val *= scale[(size_t)rr * sstride];
                    size_t o = (size_t)rr * ldc + col;
                    if (accum) C[o] += val; else C[o] = val;
                }
            }
        }
    } else {
        #pragma unroll
        for (int i = 0; i < 2; ++i) {
            #pragma unroll
            for (int j = 0; j < 2; ++j) {
                int col = colBase + wc + j * 16 + (lane & 15);
                if (col >= N) continue;
                int rbase = rowBase + wr + i * 16 + (lane >> 4) * 8;
                float bv = bias ? bias[col] : 0.0f;
                #pragma unroll
                for (int v = 0; v < 8; ++v) {
                    int rr = rbase + v;
                    if (rr >= M) continue;
                    float val = acc[i][j][v] + bv;
                    if (scale) val *= scale[(size_t)rr * sstride];
                    size_t o = (size_t)rr * ldc + col;
                    if (accum) C[o] += val; else C[o] = val;
                }
            }
        }
    }
}

// =====================================================================
// LayerNorm over last dim n (n <= 256), one row per block.
// =====================================================================
__launch_bounds__(256)
__global__ void ln_kernel(const float* __restrict__ x, float* __restrict__ o,
                          const float* __restrict__ s, const float* __restrict__ b, int n)
{
    const int row = blockIdx.x;
    const int t   = threadIdx.x;
    float v = (t < n) ? x[(size_t)row * n + t] : 0.0f;
    __shared__ float rs[256], rq[256];
    rs[t] = v; rq[t] = v * v;
    __syncthreads();
    for (int off = 128; off > 0; off >>= 1) {
        if (t < off) { rs[t] += rs[t + off]; rq[t] += rq[t + off]; }
        __syncthreads();
    }
    float mean = rs[0] / n;
    float var  = rq[0] / n - mean * mean;
    float inv  = rsqrtf(var + 1e-5f);
    if (t < n) o[(size_t)row * n + t] = (v - mean) * inv * s[t] + b[t];
}

// =====================================================================
// Causal depthwise conv (width 4) + bias + SiLU.
// xz layout: row-major (M x 1024); xi = cols [0,512). Out: xc (M x 512).
// =====================================================================
__global__ void conv_silu_kernel(const float* __restrict__ xz,
                                 const float* __restrict__ cw,
                                 const float* __restrict__ cb,
                                 float* __restrict__ xc)
{
    int idx = blockIdx.x * blockDim.x + threadIdx.x;
    if (idx >= MTOK * D_INNER) return;
    int c = idx & (D_INNER - 1);
    int row = idx >> 9;          // /512
    int t = row & (TT - 1);
    float acc = cb[c];
    #pragma unroll
    for (int j = 0; j < D_CONV; ++j) {
        int tt = t - (D_CONV - 1) + j;
        if (tt >= 0) acc += cw[c * D_CONV + j] * xz[(size_t)(row - (D_CONV - 1) + j) * 1024 + c];
    }
    float sig = 1.0f / (1.0f + __expf(-acc));
    xc[idx] = acc * sig;
}

// =====================================================================
// softplus in place
// =====================================================================
__global__ void softplus_kernel(float* __restrict__ v, int n)
{
    int i = blockIdx.x * blockDim.x + threadIdx.x;
    if (i >= n) return;
    float x = v[i];
    v[i] = (x > 20.0f) ? x : log1pf(__expf(x));
}

// =====================================================================
// Selective scan: h[16] per channel in registers, B_t/C_t in LDS.
// grid = B * (D_INNER/256); block = 256. y = scan_y + xi * D_skip.
// =====================================================================
__launch_bounds__(256)
__global__ void scan_kernel(const float* __restrict__ dt,   // M x 512
                            const float* __restrict__ bc,   // M x 48 (xdb)
                            const float* __restrict__ xi,   // M x 512 (post conv+silu)
                            const float* __restrict__ Alog, // 512 x 16
                            const float* __restrict__ Dsk,  // 512
                            float* __restrict__ y)          // M x 512
{
    const int b = blockIdx.x >> 1;
    const int d = ((blockIdx.x & 1) << 8) + threadIdx.x;   // channel 0..511
    float A[D_STATE];
    #pragma unroll
    for (int s = 0; s < D_STATE; ++s) A[s] = -__expf(Alog[d * D_STATE + s]);
    const float dskip = Dsk[d];
    float h[D_STATE];
    #pragma unroll
    for (int s = 0; s < D_STATE; ++s) h[s] = 0.0f;

    __shared__ float Bs[D_STATE], Cs[D_STATE];
    for (int t = 0; t < TT; ++t) {
        const size_t row = (size_t)b * TT + t;
        const float dtv = dt[row * D_INNER + d];
        const float xv  = xi[row * D_INNER + d];
        __syncthreads();    // everyone done with previous Bs/Cs
        if (threadIdx.x < 32) {
            float v = bc[row * 48 + DT_RANK + threadIdx.x];
            if (threadIdx.x < D_STATE) Bs[threadIdx.x] = v;
            else                       Cs[threadIdx.x - D_STATE] = v;
        }
        __syncthreads();
        const float dx = dtv * xv;
        float acc = 0.0f;
        #pragma unroll
        for (int s = 0; s < D_STATE; ++s) {
            float dA = __expf(dtv * A[s]);
            h[s] = dA * h[s] + dx * Bs[s];
            acc += h[s] * Cs[s];
        }
        y[row * D_INNER + d] = acc + xv * dskip;
    }
}

// =====================================================================
// y *= silu(z);  z = xz cols [512,1024)
// =====================================================================
__global__ void gate_kernel(float* __restrict__ y, const float* __restrict__ xz)
{
    int idx = blockIdx.x * blockDim.x + threadIdx.x;
    if (idx >= MTOK * D_INNER) return;
    int c = idx & (D_INNER - 1);
    int row = idx >> 9;
    float z = xz[(size_t)row * 1024 + D_INNER + c];
    y[idx] *= z / (1.0f + __expf(-z));
}

// =====================================================================
// Router: logits -> softmax -> top-2 -> normalized combine weights (M x 4)
// =====================================================================
__launch_bounds__(256)
__global__ void router_kernel(const float* __restrict__ x,
                              const float* __restrict__ rW,
                              float* __restrict__ comb)
{
    __shared__ float w[EXPERTS * DMODEL];
    for (int i = threadIdx.x; i < EXPERTS * DMODEL; i += 256) w[i] = rW[i];
    __syncthreads();
    int row = blockIdx.x * 256 + threadIdx.x;
    if (row >= MTOK) return;
    const float* xr = x + (size_t)row * DMODEL;
    float l[EXPERTS] = {0.f, 0.f, 0.f, 0.f};
    for (int k = 0; k < DMODEL; ++k) {
        float xv = xr[k];
        #pragma unroll
        for (int e = 0; e < EXPERTS; ++e) l[e] += xv * w[e * DMODEL + k];
    }
    float mx = l[0];
    #pragma unroll
    for (int e = 1; e < EXPERTS; ++e) mx = fmaxf(mx, l[e]);
    float p[EXPERTS], sum = 0.0f;
    #pragma unroll
    for (int e = 0; e < EXPERTS; ++e) { p[e] = __expf(l[e] - mx); sum += p[e]; }
    #pragma unroll
    for (int e = 0; e < EXPERTS; ++e) p[e] /= sum;
    // top-2 (ties -> lower index first, matching lax.top_k)
    int i0 = 0;
    #pragma unroll
    for (int e = 1; e < EXPERTS; ++e) if (p[e] > p[i0]) i0 = e;
    int i1 = (i0 == 0) ? 1 : 0;
    #pragma unroll
    for (int e = 0; e < EXPERTS; ++e) if (e != i0 && p[e] > p[i1]) i1 = e;
    float s2 = p[i0] + p[i1];
    #pragma unroll
    for (int e = 0; e < EXPERTS; ++e) {
        float c = (e == i0) ? p[i0] / s2 : ((e == i1) ? p[i1] / s2 : 0.0f);
        comb[(size_t)row * EXPERTS + e] = c;
    }
}

// =====================================================================
// exact GELU in place
// =====================================================================
__global__ void gelu_kernel(float* __restrict__ v, int n)
{
    int i = blockIdx.x * blockDim.x + threadIdx.x;
    if (i >= n) return;
    float x = v[i];
    v[i] = 0.5f * x * (1.0f + erff(x * 0.70710678118654752f));
}

// =====================================================================
// host orchestration
// =====================================================================
static inline void launch_gemm(const float* A, int lda, const float* W, int ldw,
                               const float* bias, const float* scale, int ss,
                               float* C, int ldc, int N, int Kd, int accum,
                               hipStream_t stream)
{
    dim3 grid((N + 63) / 64, (MTOK + 127) / 128);
    gemm_bf16_wmma<<<grid, dim3(256), 0, stream>>>(A, lda, W, ldw, bias, scale, ss,
                                                   C, ldc, MTOK, N, Kd, accum);
}

extern "C" void kernel_launch(void* const* d_in, const int* in_sizes, int n_in,
                              void* d_out, int out_size, void* d_ws, size_t ws_size,
                              hipStream_t stream)
{
    // inputs in setup_inputs() dict order
    const float* z_t      = (const float*)d_in[0];
    const float* in_W     = (const float*)d_in[1];
    const float* in_b     = (const float*)d_in[2];
    const float* mn_s     = (const float*)d_in[3];
    const float* mn_b     = (const float*)d_in[4];
    const float* W_in     = (const float*)d_in[5];
    const float* conv_w   = (const float*)d_in[6];
    const float* conv_b   = (const float*)d_in[7];
    const float* W_x      = (const float*)d_in[8];
    const float* W_dt     = (const float*)d_in[9];
    const float* b_dt     = (const float*)d_in[10];
    const float* A_log    = (const float*)d_in[11];
    const float* D_skip   = (const float*)d_in[12];
    const float* W_out    = (const float*)d_in[13];
    const float* on_s     = (const float*)d_in[14];
    const float* on_b     = (const float*)d_in[15];
    const float* router_W = (const float*)d_in[16];
    const float* e_W1     = (const float*)d_in[17];
    const float* e_b1     = (const float*)d_in[18];
    const float* e_W2     = (const float*)d_in[19];
    const float* e_b2     = (const float*)d_in[20];
    const float* out_W    = (const float*)d_in[21];
    const float* out_b    = (const float*)d_in[22];
    const float* fn_s     = (const float*)d_in[23];
    const float* fn_b     = (const float*)d_in[24];
    (void)in_sizes; (void)n_in; (void)out_size; (void)ws_size;

    // workspace carve-up (f32)
    float* ws    = (float*)d_ws;
    const size_t M = MTOK;
    float* xbuf  = ws;                         // M x 256   running residual
    float* lnx   = xbuf  + M * 256;            // M x 256   LN output
    float* xz    = lnx   + M * 256;            // M x 1024  W_in out; reused as FFN hidden
    float* xconv = xz    + M * 1024;           // M x 512   conv+silu out
    float* dtb   = xconv + M * 512;            // M x 512   dt
    float* ybuf  = dtb   + M * 512;            // M x 512   scan out
    float* xdb   = ybuf  + M * 512;            // M x 48 (alloc 64); reused as out-proj
    float* comb  = xdb   + M * 64;             // M x 4     combine weights

    const int EB = 256;
    const int nInner = MTOK * D_INNER;
    const int nFfn   = MTOK * D_FFN;

    // x = z_t @ in_W^T + in_b
    launch_gemm(z_t, D_IN, in_W, D_IN, in_b, nullptr, 0, xbuf, DMODEL, DMODEL, D_IN, 0, stream);

    for (int l = 0; l < LAYERS; ++l) {
        // ---- Mamba block ----
        ln_kernel<<<MTOK, 256, 0, stream>>>(xbuf, lnx, mn_s + l * DMODEL, mn_b + l * DMODEL, DMODEL);
        launch_gemm(lnx, DMODEL, W_in + (size_t)l * 2 * D_INNER * DMODEL, DMODEL,
                    nullptr, nullptr, 0, xz, 1024, 2 * D_INNER, DMODEL, 0, stream);
        conv_silu_kernel<<<(nInner + EB - 1) / EB, EB, 0, stream>>>(
            xz, conv_w + (size_t)l * D_INNER * D_CONV, conv_b + (size_t)l * D_INNER, xconv);
        launch_gemm(xconv, D_INNER, W_x + (size_t)l * 48 * D_INNER, D_INNER,
                    nullptr, nullptr, 0, xdb, 48, 48, D_INNER, 0, stream);
        launch_gemm(xdb, 48, W_dt + (size_t)l * D_INNER * DT_RANK, DT_RANK,
                    b_dt + (size_t)l * D_INNER, nullptr, 0, dtb, D_INNER, D_INNER, DT_RANK, 0, stream);
        softplus_kernel<<<(nInner + EB - 1) / EB, EB, 0, stream>>>(dtb, nInner);
        scan_kernel<<<BB * (D_INNER / 256), 256, 0, stream>>>(
            dtb, xdb, xconv, A_log + (size_t)l * D_INNER * D_STATE, D_skip + (size_t)l * D_INNER, ybuf);
        gate_kernel<<<(nInner + EB - 1) / EB, EB, 0, stream>>>(ybuf, xz);
        // x += y @ W_out^T  (residual folded into GEMM epilogue)
        launch_gemm(ybuf, D_INNER, W_out + (size_t)l * DMODEL * D_INNER, D_INNER,
                    nullptr, nullptr, 0, xbuf, DMODEL, DMODEL, D_INNER, 1, stream);

        // ---- MoE block ----
        ln_kernel<<<MTOK, 256, 0, stream>>>(xbuf, lnx, on_s + l * DMODEL, on_b + l * DMODEL, DMODEL);
        router_kernel<<<MTOK / 256, 256, 0, stream>>>(lnx, router_W + (size_t)l * EXPERTS * DMODEL, comb);
        for (int e = 0; e < EXPERTS; ++e) {
            const size_t le = (size_t)l * EXPERTS + e;
            launch_gemm(lnx, DMODEL, e_W1 + le * D_FFN * DMODEL, DMODEL,
                        e_b1 + le * D_FFN, nullptr, 0, xz /*hidden*/, D_FFN, D_FFN, DMODEL, 0, stream);
            gelu_kernel<<<(nFfn + EB - 1) / EB, EB, 0, stream>>>(xz, nFfn);
            // x += comb[:,e] * (h @ W2^T + b2)
            launch_gemm(xz, D_FFN, e_W2 + le * DMODEL * D_FFN, D_FFN,
                        e_b2 + le * DMODEL, comb + e, EXPERTS, xbuf, DMODEL, DMODEL, D_FFN, 1, stream);
        }
    }

    // out = x @ out_W^T + out_b ; final LN -> d_out
    launch_gemm(xbuf, DMODEL, out_W, DMODEL, out_b, nullptr, 0, xdb /*reuse*/, D_OUT, D_OUT, DMODEL, 0, stream);
    ln_kernel<<<MTOK, 256, 0, stream>>>(xdb, (float*)d_out, fn_s, fn_b, D_OUT);
}
